// Network_15650860827150
// MI455X (gfx1250) — matrix operand order repaired and checked by reading, not compiled, not run on hardware
//
#include <hip/hip_runtime.h>
#include <math.h>

#define NN   8000
#define NE   128000
#define DIMD 256
#define NG   64

typedef __attribute__((ext_vector_type(16))) __bf16 bf16x16;
typedef __attribute__((ext_vector_type(8)))  float  f32x8;

union FragBF { bf16x16 v; uint4 q[2]; };

__device__ __forceinline__ unsigned short f2bf(float f) {
    unsigned int u = __float_as_uint(f);
    u = u + 0x7FFFu + ((u >> 16) & 1u);      // round-to-nearest-even
    return (unsigned short)(u >> 16);
}

// ---------------- elementwise / small kernels ----------------

__global__ void k_fill_f(float* p, float v, int n) {
    int i = blockIdx.x * 256 + threadIdx.x;
    if (i < n) p[i] = v;
}
__global__ void k_fill_i(int* p, int v, int n) {
    int i = blockIdx.x * 256 + threadIdx.x;
    if (i < n) p[i] = v;
}
__global__ void k_relu(float* p, int n) {
    int i = blockIdx.x * 256 + threadIdx.x;
    if (i < n) p[i] = fmaxf(p[i], 0.f);
}
__global__ void k_residual(float* x, const float* a, const float* b, int n) {
    int i = blockIdx.x * 256 + threadIdx.x;
    if (i < n) x[i] += 0.5f * (a[i] + b[i]);
}
__global__ void k_conv(const float* __restrict__ in, unsigned short* __restrict__ out, int n) {
    int i = blockIdx.x * 256 + threadIdx.x;
    if (i < n) out[i] = f2bf(in[i]);
}
// W[e][k][n] (f32, row-major) -> out[e][n][k] (bf16 transposed)
__global__ void k_wconv(const float* __restrict__ W, unsigned short* __restrict__ out,
                        int K, int N, int total) {
    int i = blockIdx.x * 256 + threadIdx.x;
    if (i >= total) return;
    int kn = K * N;
    int e = i / kn;
    int rem = i - e * kn;
    int k = rem / N;
    int n = rem - k * N;
    out[((size_t)e * N + n) * K + k] = f2bf(W[i]);
}
__global__ void k_softmax4(const float* __restrict__ a, float* __restrict__ w, int rows) {
    int r = blockIdx.x * blockDim.x + threadIdx.x;
    if (r >= rows) return;
    float a0 = a[r*4+0], a1 = a[r*4+1], a2 = a[r*4+2], a3 = a[r*4+3];
    float m  = fmaxf(fmaxf(a0, a1), fmaxf(a2, a3));
    float e0 = __expf(a0-m), e1 = __expf(a1-m), e2 = __expf(a2-m), e3 = __expf(a3-m);
    float inv = 1.f / (e0 + e1 + e2 + e3);
    w[r*4+0] = e0*inv; w[r*4+1] = e1*inv; w[r*4+2] = e2*inv; w[r*4+3] = e3*inv;
}

// ---------------- CSR construction ----------------

__global__ void k_count(const int* __restrict__ dst, int* __restrict__ degi, int nE) {
    int e = blockIdx.x * 256 + threadIdx.x;
    if (e < nE) atomicAdd(&degi[dst[e]], 1);
}
__global__ void k_scan(const int* __restrict__ degi, int* __restrict__ rowptr,
                       float* __restrict__ invdeg, int n) {
    if (blockIdx.x == 0 && threadIdx.x == 0) {
        int acc = 0;
        for (int i = 0; i < n; ++i) {
            rowptr[i] = acc;
            acc += degi[i];
            invdeg[i] = 1.f / fmaxf((float)degi[i], 1.f);
        }
        rowptr[n] = acc;
    }
}
__global__ void k_copy_i(const int* __restrict__ a, int* __restrict__ b, int n) {
    int i = blockIdx.x * 256 + threadIdx.x;
    if (i < n) b[i] = a[i];
}
__global__ void k_build(const int* __restrict__ src, const int* __restrict__ dst,
                        int* __restrict__ cursor, int* __restrict__ eperm, int nE) {
    int e = blockIdx.x * 256 + threadIdx.x;
    if (e < nE) {
        int p = atomicAdd(&cursor[dst[e]], 1);
        eperm[p] = src[e];
    }
}

// ---------------- aggregations (CSR gather, no atomics) ----------------

// aggbf[n][d] = bf16( (1/deg[n]) * sum_{e in N(n)} x[eperm[e]][d] )
__global__ __launch_bounds__(DIMD) void k_agg_bf(const float* __restrict__ x,
        const int* __restrict__ rowptr, const int* __restrict__ eperm,
        const float* __restrict__ invdeg, unsigned short* __restrict__ aggbf) {
    int n = blockIdx.x, d = threadIdx.x;
    int beg = rowptr[n], end = rowptr[n+1];
    float acc = 0.f;
    for (int e = beg; e < end; ++e) acc += x[(size_t)eperm[e] * DIMD + d];
    aggbf[(size_t)n * DIMD + d] = f2bf(acc * invdeg[n]);
}

// mid = w0*xn + w1*max + w2*mean + w3*sum   (max := 0 when no neighbors)
__global__ __launch_bounds__(DIMD) void k_middle(const float* __restrict__ xn,
        const int* __restrict__ rowptr, const int* __restrict__ eperm,
        const float* __restrict__ invdeg, const float* __restrict__ wrow,
        float* __restrict__ out) {
    int n = blockIdx.x, d = threadIdx.x;
    int beg = rowptr[n], end = rowptr[n+1];
    float s = 0.f, mx = -INFINITY;
    for (int e = beg; e < end; ++e) {
        float v = xn[(size_t)eperm[e] * DIMD + d];
        s += v;
        mx = fmaxf(mx, v);
    }
    if (end == beg) mx = 0.f;
    float w0 = wrow[0], w1 = wrow[1], w2 = wrow[2], w3 = wrow[3];
    size_t i = (size_t)n * DIMD + d;
    out[i] = w0 * xn[i] + w1 * mx + w2 * s * invdeg[n] + w3 * s;
}

// ---------------- fused WMMA mixed-op GEMM ----------------
// S (store/accum) = w1*Xid + w2*(A1 @ B1t^T + bias1) + w3*(A2 @ B2t^T + bias2)
// A*: M x K bf16 row-major.  B*t: N x K bf16 (i.e. W^T).  M=8000, N=256.
// Block = 4 waves; wave w owns rows [m0,m0+16) x cols [w*64, w*64+64).
__global__ __launch_bounds__(128) void k_gemm_mixed(
        const unsigned short* __restrict__ A1, const unsigned short* __restrict__ B1t,
        const float* __restrict__ bias1,
        const unsigned short* __restrict__ A2, const unsigned short* __restrict__ B2t,
        const float* __restrict__ bias2,
        const float* __restrict__ Xid, const float* __restrict__ wrow,
        float* __restrict__ S, int K, int accum) {
    const int lane = threadIdx.x & 31;
    const int n0   = (threadIdx.x >> 5) * 64;
    const int m0   = blockIdx.x * 16;
    const int l15  = lane & 15;
    const int hi   = lane >> 4;

    f32x8 acc1[4], acc2[4];
#pragma unroll
    for (int j = 0; j < 4; ++j)
#pragma unroll
        for (int r = 0; r < 8; ++r) { acc1[j][r] = 0.f; acc2[j][r] = 0.f; }

    const size_t arow = (size_t)(m0 + l15) * K;   // A: lane = row (both halves)
    for (int kk = 0; kk < K; kk += 32) {
        FragBF a1;
        const unsigned short* pa = A1 + arow + kk + hi * 8;   // K {0..7,16..23} / {8..15,24..31}
        a1.q[0] = *(const uint4*)pa;
        a1.q[1] = *(const uint4*)(pa + 16);
#pragma unroll
        for (int j = 0; j < 4; ++j) {
            FragBF b;   // B: lane = column; K {0..7,16..23} / {8..15,24..31}
            const unsigned short* pb = B1t + (size_t)(n0 + j*16 + l15) * K + kk + hi * 8;
            b.q[0] = *(const uint4*)pb;
            b.q[1] = *(const uint4*)(pb + 16);
            acc1[j] = __builtin_amdgcn_wmma_f32_16x16x32_bf16(
                false, a1.v, false, b.v, (short)0, acc1[j], false, false);
        }
        if (A2) {
            FragBF a2;
            const unsigned short* pa2 = A2 + arow + kk + hi * 8;
            a2.q[0] = *(const uint4*)pa2;
            a2.q[1] = *(const uint4*)(pa2 + 16);
#pragma unroll
            for (int j = 0; j < 4; ++j) {
                FragBF b;
                const unsigned short* pb = B2t + (size_t)(n0 + j*16 + l15) * K + kk + hi * 8;
                b.q[0] = *(const uint4*)pb;
                b.q[1] = *(const uint4*)(pb + 16);
                acc2[j] = __builtin_amdgcn_wmma_f32_16x16x32_bf16(
                    false, a2.v, false, b.v, (short)0, acc2[j], false, false);
            }
        }
    }

    float w1 = 0.f, w2 = 1.f, w3 = 0.f;
    if (wrow) { w1 = wrow[1]; w2 = wrow[2]; w3 = wrow[3]; }
#pragma unroll
    for (int j = 0; j < 4; ++j) {
        const int col = n0 + j * 16 + l15;
        const float b1v = bias1 ? bias1[col] : 0.f;
        const float b2v = bias2 ? bias2[col] : 0.f;
#pragma unroll
        for (int r = 0; r < 8; ++r) {
            const int row = m0 + r + 8 * hi;                  // C/D layout: VGPR r -> M=r / M=r+8
            const size_t idx = (size_t)row * DIMD + col;
            float v = w2 * (acc1[j][r] + b1v);
            if (A2)  v += w3 * (acc2[j][r] + b2v);
            if (Xid) v += w1 * Xid[idx];
            if (accum) S[idx] += v; else S[idx] = v;
        }
    }
}

// ---------------- readout ----------------

__global__ __launch_bounds__(DIMD) void k_pool(const float* __restrict__ x,
        const int* __restrict__ gid, float* __restrict__ hg, float* __restrict__ cnt) {
    int n = blockIdx.x, d = threadIdx.x;
    int g = gid[n];
    atomicAdd(&hg[(size_t)g * DIMD + d], x[(size_t)n * DIMD + d]);
    if (d == 0) atomicAdd(&cnt[g], 1.f);
}

__global__ __launch_bounds__(128) void k_mlp(const float* __restrict__ hg,
        const float* __restrict__ cnt,
        const float* __restrict__ C1, const float* __restrict__ c1b,
        const float* __restrict__ C2, const float* __restrict__ c2b,
        const float* __restrict__ C3, const float* __restrict__ c3b,
        float* __restrict__ out) {
    __shared__ float xg[256];
    __shared__ float y1[128];
    __shared__ float y2[64];
    int g = blockIdx.x, t = threadIdx.x;
    float inv = 1.f / fmaxf(cnt[g], 1.f);
    xg[t]       = hg[(size_t)g * 256 + t] * inv;
    xg[t + 128] = hg[(size_t)g * 256 + t + 128] * inv;
    __syncthreads();
    float a1 = c1b[t];
    for (int k = 0; k < 256; ++k) a1 += xg[k] * C1[k * 128 + t];
    y1[t] = fmaxf(a1, 0.f);
    __syncthreads();
    if (t < 64) {
        float a2 = c2b[t];
        for (int k = 0; k < 128; ++k) a2 += y1[k] * C2[k * 64 + t];
        y2[t] = fmaxf(a2, 0.f);
    }
    __syncthreads();
    if (t < 10) {
        float a3 = c3b[t];
        for (int k = 0; k < 64; ++k) a3 += y2[k] * C3[k * 10 + t];
        out[g * 10 + t] = a3;
    }
}

// ---------------- host orchestration ----------------

extern "C" void kernel_launch(void* const* d_in, const int* in_sizes, int n_in,
                              void* d_out, int out_size, void* d_ws, size_t ws_size,
                              hipStream_t stream) {
    (void)in_sizes; (void)n_in; (void)out_size; (void)ws_size;

    const float* h    = (const float*)d_in[0];
    const float* Wemb = (const float*)d_in[1];
    const float* bemb = (const float*)d_in[2];
    const float* WfDf = (const float*)d_in[3];
    const float* bfDf = (const float*)d_in[4];
    const float* WfSf = (const float*)d_in[5];
    const float* bfSf = (const float*)d_in[6];
    const float* WlDf = (const float*)d_in[7];
    const float* blDf = (const float*)d_in[8];
    const float* WlSf = (const float*)d_in[9];
    const float* blSf = (const float*)d_in[10];
    const float* aF   = (const float*)d_in[11];
    const float* aM   = (const float*)d_in[12];
    const float* aL   = (const float*)d_in[13];
    const float* C1   = (const float*)d_in[14];
    const float* c1b  = (const float*)d_in[15];
    const float* C2   = (const float*)d_in[16];
    const float* c2b  = (const float*)d_in[17];
    const float* C3   = (const float*)d_in[18];
    const float* c3b  = (const float*)d_in[19];
    const int*   src  = (const int*)d_in[20];
    const int*   dstE = (const int*)d_in[21];
    const int*   gid  = (const int*)d_in[22];

    char* base = (char*)d_ws;
    size_t off = 0;
    auto take = [&](size_t bytes) -> char* {
        char* p = base + off;
        off = (off + bytes + 255) & ~(size_t)255;
        return p;
    };
    float* X    = (float*)take(sizeof(float) * NN * DIMD);
    float* S1   = (float*)take(sizeof(float) * NN * DIMD);
    float* S2   = (float*)take(sizeof(float) * NN * DIMD);
    float* MID0 = (float*)take(sizeof(float) * NN * DIMD);
    float* MID1 = (float*)take(sizeof(float) * NN * DIMD);
    unsigned short* XBF  = (unsigned short*)take(2ull * NN * DIMD);
    unsigned short* AGBF = (unsigned short*)take(2ull * NN * DIMD);
    unsigned short* HBF  = (unsigned short*)take(2ull * NN * 128);
    unsigned short* WET  = (unsigned short*)take(2ull * 128 * 256);
    unsigned short* WFD  = (unsigned short*)take(2ull * 6  * 65536);
    unsigned short* WFS  = (unsigned short*)take(2ull * 6  * 65536);
    unsigned short* WLD  = (unsigned short*)take(2ull * 10 * 65536);
    unsigned short* WLS  = (unsigned short*)take(2ull * 10 * 65536);
    float* WF   = (float*)take(sizeof(float) * 24);
    float* WM   = (float*)take(sizeof(float) * 16);
    float* WL   = (float*)take(sizeof(float) * 40);
    int* DEGI   = (int*)take(sizeof(int) * NN);
    int* ROWPTR = (int*)take(sizeof(int) * (NN + 1));
    int* CURSOR = (int*)take(sizeof(int) * NN);
    int* EPRM   = (int*)take(sizeof(int) * NE);   // (renamed: EPERM is a glibc errno macro)
    float* INVD = (float*)take(sizeof(float) * NN);
    float* HG   = (float*)take(sizeof(float) * NG * DIMD);
    float* CNT  = (float*)take(sizeof(float) * NG);

    const int ND = NN * DIMD;

    // architecture-weight softmaxes
    k_softmax4<<<1, 64, 0, stream>>>(aF, WF, 6);
    k_softmax4<<<1, 64, 0, stream>>>(aM, WM, 4);
    k_softmax4<<<1, 64, 0, stream>>>(aL, WL, 10);

    // CSR over dst
    k_fill_i<<<(NN + 255) / 256, 256, 0, stream>>>(DEGI, 0, NN);
    k_count<<<(NE + 255) / 256, 256, 0, stream>>>(dstE, DEGI, NE);
    k_scan<<<1, 1, 0, stream>>>(DEGI, ROWPTR, INVD, NN);
    k_copy_i<<<(NN + 255) / 256, 256, 0, stream>>>(ROWPTR, CURSOR, NN);
    k_build<<<(NE + 255) / 256, 256, 0, stream>>>(src, dstE, CURSOR, EPRM, NE);

    // weights -> bf16 transposed (N x K)
    k_wconv<<<(128 * 256 + 255) / 256, 256, 0, stream>>>(Wemb, WET, 128, 256, 128 * 256);
    k_wconv<<<(6  * 65536 + 255) / 256, 256, 0, stream>>>(WfDf, WFD, 256, 256, 6  * 65536);
    k_wconv<<<(6  * 65536 + 255) / 256, 256, 0, stream>>>(WfSf, WFS, 256, 256, 6  * 65536);
    k_wconv<<<(10 * 65536 + 255) / 256, 256, 0, stream>>>(WlDf, WLD, 256, 256, 10 * 65536);
    k_wconv<<<(10 * 65536 + 255) / 256, 256, 0, stream>>>(WlSf, WLS, 256, 256, 10 * 65536);

    // embedding: X = h @ W_emb + b_emb
    k_conv<<<(NN * 128 + 255) / 256, 256, 0, stream>>>(h, HBF, NN * 128);
    k_gemm_mixed<<<NN / 16, 128, 0, stream>>>(HBF, WET, bemb, nullptr, nullptr, nullptr,
                                              nullptr, nullptr, X, 128, 0);

    auto prep = [&](const float* xj) {
        k_conv<<<(ND + 255) / 256, 256, 0, stream>>>(xj, XBF, ND);
        k_agg_bf<<<NN, DIMD, 0, stream>>>(xj, ROWPTR, EPRM, INVD, AGBF);
    };
    auto mixedg = [&](const float* xj, int stage, int l, int e, float* S, int accum) {
        const unsigned short *Wd, *Ws;
        const float *bd, *bs, *wr;
        if (stage == 0) {
            int idx = l * 3 + e;
            Wd = WFD + (size_t)idx * 65536; Ws = WFS + (size_t)idx * 65536;
            bd = bfDf + idx * 256;          bs = bfSf + idx * 256;
            wr = WF + idx * 4;
        } else {
            int idx = l * 5 + e;
            Wd = WLD + (size_t)idx * 65536; Ws = WLS + (size_t)idx * 65536;
            bd = blDf + idx * 256;          bs = blSf + idx * 256;
            wr = WL + idx * 4;
        }
        k_gemm_mixed<<<NN / 16, 128, 0, stream>>>(XBF, Wd, bd, AGBF, Ws, bs, xj, wr, S, 256, accum);
    };

    for (int l = 0; l < 2; ++l) {
        // first-stage DAG: s1 = mixed(x,e0); s2 = mixed(x,e1) + mixed(relu(s1),e2)
        prep(X);
        mixedg(X, 0, l, 0, S1, 0);
        mixedg(X, 0, l, 1, S2, 0);
        k_relu<<<(ND + 255) / 256, 256, 0, stream>>>(S1, ND);
        prep(S1);
        mixedg(S1, 0, l, 2, S2, 1);
        k_relu<<<(ND + 255) / 256, 256, 0, stream>>>(S2, ND);
        // middle stage mixtures
        k_middle<<<NN, DIMD, 0, stream>>>(S1, ROWPTR, EPRM, INVD, WM + (l * 2 + 0) * 4, MID0);
        k_middle<<<NN, DIMD, 0, stream>>>(S2, ROWPTR, EPRM, INVD, WM + (l * 2 + 1) * 4, MID1);
        // last-stage DAG: last0 -> S1, last1 -> S2
        prep(MID0);
        mixedg(MID0, 1, l, 0, S1, 0);
        mixedg(MID0, 1, l, 2, S2, 0);
        prep(MID1);
        mixedg(MID1, 1, l, 1, S1, 1);
        mixedg(MID1, 1, l, 3, S2, 1);
        k_relu<<<(ND + 255) / 256, 256, 0, stream>>>(S1, ND);
        prep(S1);
        mixedg(S1, 1, l, 4, S2, 1);
        k_relu<<<(ND + 255) / 256, 256, 0, stream>>>(S2, ND);
        // residual: x += (last0 + last1)/2
        k_residual<<<(ND + 255) / 256, 256, 0, stream>>>(X, S1, S2, ND);
    }

    // per-graph mean readout + MLP
    k_fill_f<<<(NG * DIMD + 255) / 256, 256, 0, stream>>>(HG, 0.f, NG * DIMD);
    k_fill_f<<<1, 64, 0, stream>>>(CNT, 0.f, NG);
    k_pool<<<NN, DIMD, 0, stream>>>(X, gid, HG, CNT);
    k_mlp<<<NG, 128, 0, stream>>>(HG, CNT, C1, c1b, C2, c2b, C3, c3b, (float*)d_out);
}
